// InvConvGamma_81578608820556
// MI455X (gfx1250) — matrix operand order repaired and checked by reading, not compiled, not run on hardware
//
#include <hip/hip_runtime.h>

// ---------------------------------------------------------------------------
// InvGammatone depthwise FIR (K~8390 taps, NumCh=65) + per-channel shift +
// channel-sum, as banded-Toeplitz GEMM on CDNA5 WMMA (wave32):
//
//   out[b,t] = (1/fs) * sum_c sum_j x[b,c, t + s_c + j] * w[c,j]
//   s_c = gather_idx[c,0] - (K-1);  |z|cos(arg z) == Re z;  x zero outside [0,T)
//
// Per-wave dual 16x16 tiles:  D[m,n] = out[t0w + 256*tile + 16m + n]
//   A[m,k] = x[t0w + s_c + k0 + 16m + k]        (Toeplitz signal slice, LDS)
//   B[k,n] = w[k0 + k - n], 0 outside [0,K)     (banded weights, LDS)
// B fragments are shared by both tiles -> 3 ds_load_b128 per WMMA.
//
// The 16 shifted band rows ws2[r][k] = wpad[16 + k0 + k - (15-r)] are produced
// by ONE TDM tensor_load_to_lds (iterate_enable, 16 iterations, global step =
// 1 element, LDS step = WS2S) from a one-time zero-padded bf16 weight array in
// d_ws. Channel groups across grid.z write unscaled partials; a reduce kernel
// sums groups and applies 1/fs (deterministic, no float atomics).
// ---------------------------------------------------------------------------

typedef __attribute__((ext_vector_type(16))) __bf16        v16bf;
typedef __attribute__((ext_vector_type(8)))  float         v8f;
typedef __attribute__((ext_vector_type(4)))  unsigned int  v4u;
typedef __attribute__((ext_vector_type(4)))  unsigned int  u32x4;
typedef __attribute__((ext_vector_type(8)))  int           i32x8;
typedef __attribute__((ext_vector_type(4)))  int           i32x4;

#define KC      256                 // reduction rows staged per LDS pass
#define TILE_T  2048                // outputs per block (4 waves x 2 tiles x 256)
#define NWAVES  4
#define BLOCK   (NWAVES * 32)       // wave32
#define WX      (TILE_T + KC)       // staged signal window (elements)
#define WS2S    (KC + 8)            // padded row stride of band matrix (bank spread)
#define WPAD_F  16                  // front zero-pad of bf16 weight array
#define MAXCG   5                   // channel groups (grid.z)

__device__ __forceinline__ unsigned short f2bf(float f) {
    unsigned int u = __float_as_uint(f);
    u += 0x7FFFu + ((u >> 16) & 1u);          // round-to-nearest-even
    return (unsigned short)(u >> 16);
}

union FragBF { v4u q[2]; v16bf v; };

// ---- one-time: zero-padded bf16 weights: wpad[c][WPAD_F + j] = bf16(w[c][j])
__global__ void prep_wpad(const float* __restrict__ weight, unsigned short* __restrict__ wpad,
                          int K, int wpadLen) {
    const int c = blockIdx.x;
    const float* wrow = weight + (size_t)c * (size_t)K;
    unsigned short* prow = wpad + (size_t)c * (size_t)wpadLen;
    for (int i = threadIdx.x; i < wpadLen; i += blockDim.x) {
        int j = i - WPAD_F;
        prow[i] = (j >= 0 && j < K) ? f2bf(wrow[j]) : (unsigned short)0;
    }
}

__global__ __launch_bounds__(BLOCK)
void invconv_gamma_wmma(const float* __restrict__ inp,             // [B, 2*NumCh, T]
                        const float* __restrict__ weight,          // [NumCh, 1, K]
                        const unsigned short* __restrict__ wpad,   // bf16 padded (or null)
                        const int*   __restrict__ gidx,            // [NumCh, T]
                        float*       __restrict__ partial,         // [CG, B, T] unscaled
                        int NumCh, int T, int K, int chPer, int wpadLen)
{
    __shared__ __align__(16) unsigned short xs [WX];          // bf16 signal window
    __shared__ __align__(16) unsigned short ws2[16 * WS2S];   // 16 shifted band rows

    const int tid   = threadIdx.x;
    const int lane  = tid & 31;
    const int wv    = tid >> 5;
    const int mn    = lane & 15;         // A row m / B column n / D column N
    const int hi    = lane >> 4;
    const int t0blk = blockIdx.x * TILE_T;
    const int bb    = blockIdx.y;
    const int cg    = blockIdx.z;

    // Fragment element offsets (16B-aligned ds_load_b128):
    //  A 16x32 bf16: lanes0-15 K{0..7},{16..23}; lanes16-31 K{8..15},{24..31}
    //  B 32x16 bf16: lanes0-15 K{0..7},{16..23}; lanes16-31 K{8..15},{24..31}
    // TDM iteration r lands at row r = 15 - n  ->  bBase uses (15 - mn).
    const int aBase0 = 512 * wv + 16 * mn + 8 * hi;      // tile 0 (+256 for tile 1)
    const int bBase  = (15 - mn) * WS2S + 8 * hi;

    const v4u* xq = reinterpret_cast<const v4u*>(xs);
    const v4u* wq = reinterpret_cast<const v4u*>(ws2);
    const unsigned int ldsWs2 = (unsigned int)(size_t)(&ws2[0]);   // LDS byte offset

    v8f acc0 = {}, acc1 = {};
    const int KB   = K + 15;
    const int cBeg = cg * chPer;
    const int cEnd = (cBeg + chPer < NumCh) ? cBeg + chPer : NumCh;

    for (int c = cBeg; c < cEnd; ++c) {
        const float* __restrict__ xrow = inp + ((size_t)bb * (2 * NumCh) + c) * (size_t)T;
        const int shift = gidx[(size_t)c * (size_t)T] - (K - 1);   // s_c

        for (int k0 = 0; k0 < KB; k0 += KC) {
            const int gbase = t0blk + shift + k0;

            __syncthreads();   // previous chunk's fragment reads complete

            // ---- stage bf16 signal window (zero-padded, f32 -> bf16) ----
#pragma unroll
            for (int it = 0; it < WX / BLOCK; ++it) {
                int i = tid + it * BLOCK;
                int g = gbase + i;
                float v = (g >= 0 && g < T) ? xrow[g] : 0.0f;
                xs[i] = f2bf(v);
            }

            // ---- stage 16 shifted band rows of weights ----
            if (wpad) {
                const unsigned short* prow = wpad + (size_t)c * (size_t)wpadLen;
#if __has_builtin(__builtin_amdgcn_tensor_load_to_lds) && __has_builtin(__builtin_amdgcn_s_wait_tensorcnt)
                if (tid < 32) {   // wave 0 issues one TDM op (EXEC ignored by TDM)
                    unsigned long long ga =
                        (unsigned long long)(size_t)(prow + (WPAD_F + k0 - 15));
                    u32x4 g0;
                    g0[0] = 1u;                                   // count=1
                    g0[1] = ldsWs2;                               // lds_addr (bytes)
                    g0[2] = (unsigned int)(ga & 0xFFFFFFFFu);     // global_addr[31:0]
                    g0[3] = (unsigned int)((ga >> 32) & 0x1FFFFFFu) | (2u << 30); // [56:32]|type=2
                    i32x8 g1;
                    g1[0] = (1 << 16) | (1 << 19);                // data_size=2B, iterate_enable
                    g1[1] = 0;                                    // tensor_dim0 lo16<<16 (2^30 -> 0)
                    g1[2] = 0x4000 | (1 << 16);                   // tensor_dim0 hi16, tensor_dim1=1
                    g1[3] = (KC << 16);                           // tile_dim0 = KC
                    g1[4] = 1;                                    // tile_dim1=1, tile_dim2=0
                    g1[5] = (int)(1u << 30);                      // tensor_dim0_stride lo32
                    g1[6] = 0;                                    // stride hi16 | dim1_stride lo16
                    g1[7] = 0;
                    i32x4 g2;
                    g2[0] = 1;                                    // tensor_dim2 (unused)
                    g2[1] = WS2S;                                 // lds_addr_increment (elements)
                    g2[2] = 1;                                    // global_addr_increment = 1 elem
                    g2[3] = (15 << 16);                           // iterate_count = 15 (16 iters)
                    i32x4 g3 = {0, 0, 0, 0};
  #if defined(__clang_major__) && (__clang_major__ >= 23)
                    i32x8 g4 = {0, 0, 0, 0, 0, 0, 0, 0};
                    __builtin_amdgcn_tensor_load_to_lds(g0, g1, g2, g3, g4, 0);
  #else
                    __builtin_amdgcn_tensor_load_to_lds(g0, g1, g2, g3, 0);
  #endif
                    __builtin_amdgcn_s_wait_tensorcnt(0);
                }
#else
                // bf16 copy fallback: row r holds wpad[WPAD_F + k0 + kk - (15-r)]
#pragma unroll
                for (int it = 0; it < (16 * KC) / BLOCK; ++it) {
                    int i  = tid + it * BLOCK;
                    int r  = i >> 8;
                    int kk = i & (KC - 1);
                    ws2[r * WS2S + kk] = prow[WPAD_F + k0 - 15 + r + kk];
                }
#endif
            } else {
                // full fallback: f32 weights, bounds-checked, converted on the fly
                const float* wrow = weight + (size_t)c * (size_t)K;
#pragma unroll
                for (int it = 0; it < (16 * KC) / BLOCK; ++it) {
                    int i  = tid + it * BLOCK;
                    int r  = i >> 8;
                    int kk = i & (KC - 1);
                    int j  = k0 + kk - (15 - r);
                    float v = (j >= 0 && j < K) ? wrow[j] : 0.0f;
                    ws2[r * WS2S + kk] = f2bf(v);
                }
            }

            // warm caches for next chunk's signal window (global_prefetch_b8)
            {
                int gpre = gbase + KC + (int)tid * 16;
                if (gpre >= 0 && gpre < T)
                    __builtin_prefetch((const void*)(xrow + gpre), 0, 1);
            }

            __syncthreads();

            // ---- 8 x (2 WMMAs sharing one B fragment): 3 ds_load_b128 / WMMA ----
#pragma unroll
            for (int kc = 0; kc < KC / 32; ++kc) {
                FragBF fb, fa0, fa1;
                int eb = bBase + 32 * kc;
                fb.q[0] = wq[(eb >> 3)];
                fb.q[1] = wq[(eb >> 3) + 2];
                int e0 = aBase0 + 32 * kc;
                fa0.q[0] = xq[(e0 >> 3)];
                fa0.q[1] = xq[(e0 >> 3) + 2];
                int e1 = e0 + 256;
                fa1.q[0] = xq[(e1 >> 3)];
                fa1.q[1] = xq[(e1 >> 3) + 2];
                acc0 = __builtin_amdgcn_wmma_f32_16x16x32_bf16(
                           false, fa0.v, false, fb.v, (short)0, acc0, false, false);
                acc1 = __builtin_amdgcn_wmma_f32_16x16x32_bf16(
                           false, fa1.v, false, fb.v, (short)0, acc1, false, false);
            }
        }
    }

    // ---- store D tiles: VGPR r -> (M = r + 8*hi, N = mn) ----
    float* __restrict__ prow = partial + ((size_t)cg * gridDim.y + bb) * (size_t)T;
    const int t0w = t0blk + 512 * wv;
#pragma unroll
    for (int r = 0; r < 8; ++r) {
        int trow = 16 * (r + 8 * hi) + mn;
        int t0 = t0w + trow;
        if (t0 < T)       prow[t0]       = acc0[r];
        int t1 = t0w + 256 + trow;
        if (t1 < T)       prow[t1]       = acc1[r];
    }
}

// ---- sum channel-group partials, apply 1/fs (deterministic order) ----
__global__ void reduce_groups(const float* __restrict__ partial, float* __restrict__ out,
                              const int* __restrict__ fsPtr, int BT, int CG) {
    int i = blockIdx.x * blockDim.x + threadIdx.x;
    if (i >= BT) return;
    float s = 0.0f;
    for (int g = 0; g < CG; ++g) s += partial[(size_t)g * BT + i];
    out[i] = s * (1.0f / (float)(*fsPtr));
}

extern "C" void kernel_launch(void* const* d_in, const int* in_sizes, int n_in,
                              void* d_out, int out_size, void* d_ws, size_t ws_size,
                              hipStream_t stream) {
    const float* inp    = (const float*)d_in[0];
    const float* weight = (const float*)d_in[1];
    const int*   gidx   = (const int*)d_in[2];
    const int*   fsPtr  = (const int*)d_in[3];
    float*       out    = (float*)d_out;

    // in_sizes[0]=2*B*NumCh*T, in_sizes[1]=NumCh*K, in_sizes[2]=NumCh*T, out=B*T
    long long numCh = (long long)in_sizes[0] / (2LL * (long long)out_size);
    if (numCh <= 0) numCh = 1;
    int T = (int)((long long)in_sizes[2] / numCh);
    int B = out_size / T;
    int K = (int)((long long)in_sizes[1] / numCh);

    int nChunks = (K + 15 + KC - 1) / KC;
    int wpadLen = nChunks * KC + 2 * WPAD_F;
    size_t wpadBytes = (((size_t)numCh * wpadLen * 2) + 255) & ~(size_t)255;

    unsigned short* wpad = nullptr;
    size_t off = 0;
    if (ws_size >= wpadBytes) { wpad = (unsigned short*)d_ws; off = wpadBytes; }

    int CG = 1;
    float* partial = out;                 // CG==1: write partials straight to out
    size_t partBytes = (size_t)MAXCG * (size_t)out_size * 4;
    if (ws_size >= off + partBytes) {
        CG = (numCh < MAXCG) ? (int)numCh : MAXCG;
        partial = (float*)((char*)d_ws + off);
    }
    int chPer = (int)((numCh + CG - 1) / CG);

    if (wpad)
        prep_wpad<<<dim3((unsigned)numCh), 256, 0, stream>>>(weight, wpad, K, wpadLen);

    dim3 grid((T + TILE_T - 1) / TILE_T, B, CG);
    invconv_gamma_wmma<<<grid, BLOCK, 0, stream>>>(inp, weight, wpad, gidx, partial,
                                                   (int)numCh, T, K, chPer, wpadLen);

    reduce_groups<<<(out_size + 255) / 256, 256, 0, stream>>>(partial, out, fsPtr,
                                                              out_size, CG);
}
